// Attention_49529562858354
// MI455X (gfx1250) — matrix operand order repaired and checked by reading, not compiled, not run on hardware
//
#include <hip/hip_runtime.h>

#define NDIM 1024
#define HEADS 16
#define DH 64
#define NTOK 2048
#define NB 4
#define ATT_SCALE 0.125f   // 1/sqrt(64)
#define NEG_BIG (-3.0e38f)

typedef __attribute__((ext_vector_type(16))) __bf16 v16bf;
typedef __attribute__((ext_vector_type(8)))  float v8f;
typedef unsigned int u32x4 __attribute__((ext_vector_type(4)));
typedef int i32x4 __attribute__((ext_vector_type(4)));
typedef int i32x8 __attribute__((ext_vector_type(8)));

union FragBF { v16bf v; uint4 q[2]; unsigned int u[8]; };
union AccF   { v8f v; float f[8]; };

static __device__ inline unsigned short f2bf(float f) {
  unsigned int x = __float_as_uint(f);
  unsigned int r = x + 0x7FFFu + ((x >> 16) & 1u);   // round-to-nearest-even
  return (unsigned short)(r >> 16);
}

// CDNA5 async global->LDS copy (tracked by ASYNCcnt). Flat addr[31:0] of a
// __shared__ pointer is the wave-relative LDS offset the instruction wants.
static __device__ inline void async_b128(unsigned int* lds, const unsigned int* g) {
  unsigned int l = (unsigned int)(unsigned long long)lds;
  asm volatile("global_load_async_to_lds_b128 %0, %1, off"
               :: "v"(l), "v"(g) : "memory");
}
static __device__ inline void wait_async0() {
  asm volatile("s_wait_asynccnt 0x0" ::: "memory");
}

// ---------------- Tensor Data Mover: 2D tile -> LDS (one descriptor) ----------------
// D# per CDNA5 ISA ch.8: group0 {count,lds_addr,global_addr,type=2},
// group1 {data_size=2B, pad 4 dwords after every 32 dwords (-> LDS row stride 36),
//         tensor dims/strides, tile dims}, groups 2/3 zero (2D tensor).
#define HAVE_TDM __has_builtin(__builtin_amdgcn_tensor_load_to_lds)

static __device__ inline void tdm_load_2d(unsigned int lds_addr,
                                          unsigned long long gaddr,
                                          unsigned int tensor_d0, unsigned int tensor_d1,
                                          unsigned int tile_d0,   unsigned int tile_d1,
                                          unsigned int stride0) {
#if HAVE_TDM
  u32x4 g0;
  g0.x = 1u;                                            // count=1, user mode
  g0.y = lds_addr;                                      // LDS byte address
  g0.z = (unsigned int)gaddr;                           // global_addr[31:0]
  g0.w = (unsigned int)((gaddr >> 32) & 0x1FFFFFFull)   // global_addr[56:32]
         | (2u << 30);                                  // type=2 ("image")
  i32x8 g1;
  g1[0] = (int)((1u << 16)      // data_size = 1 -> 2 bytes
              | (1u << 20)      // pad_enable
              | (4u << 22)      // pad_interval code 4 -> every 32 dwords (128B)
              | (3u << 25));    // pad_amount  code 3 -> 4 dwords (16B)
  g1[1] = (int)((tensor_d0 & 0xFFFFu) << 16);                         // dim0[15:0]
  g1[2] = (int)((tensor_d0 >> 16) | ((tensor_d1 & 0xFFFFu) << 16));   // dim0[31:16], dim1[15:0]
  g1[3] = (int)((tensor_d1 >> 16) | (tile_d0 << 16));                 // dim1[31:16], tile_dim0
  g1[4] = (int)(tile_d1 & 0xFFFFu);                                   // tile_dim1 (tile_dim2=0)
  g1[5] = (int)stride0;                                               // tensor_dim0_stride[31:0]
  g1[6] = 0;                                                          // stride0 hi, stride1 lo
  g1[7] = 0;
  i32x4 z4 = {0, 0, 0, 0};
#if defined(__clang_major__) && (__clang_major__ >= 23)
  i32x8 z8 = {0, 0, 0, 0, 0, 0, 0, 0};
  __builtin_amdgcn_tensor_load_to_lds(g0, g1, z4, z4, z8, 0);
#else
  __builtin_amdgcn_tensor_load_to_lds(g0, g1, z4, z4, 0);
#endif
#endif
}
static __device__ inline void wait_tensor0() {
#if HAVE_TDM
  __builtin_amdgcn_s_wait_tensorcnt((short)0);
#endif
}

static __device__ inline float half16_max(float v) {
  v = fmaxf(v, __shfl_xor(v, 1, 32));
  v = fmaxf(v, __shfl_xor(v, 2, 32));
  v = fmaxf(v, __shfl_xor(v, 4, 32));
  v = fmaxf(v, __shfl_xor(v, 8, 32));
  return v;
}
static __device__ inline float half16_sum(float v) {
  v += __shfl_xor(v, 1, 32);
  v += __shfl_xor(v, 2, 32);
  v += __shfl_xor(v, 4, 32);
  v += __shfl_xor(v, 8, 32);
  return v;
}

// ---------------- Kernel 1: LayerNorm (f32 -> bf16 packed) ----------------
__global__ __launch_bounds__(256) void ln_kernel(const float* __restrict__ x,
                                                 const float* __restrict__ g,
                                                 const float* __restrict__ b,
                                                 unsigned int* __restrict__ xn) {
  int row = blockIdx.x;                         // 0..8191
  const float* xr = x + (size_t)row * NDIM;
  __shared__ float red[256];
  int base = threadIdx.x * 4;
  float v0 = xr[base + 0], v1 = xr[base + 1], v2 = xr[base + 2], v3 = xr[base + 3];
  float s = v0 + v1 + v2 + v3;
  float s2 = v0 * v0 + v1 * v1 + v2 * v2 + v3 * v3;

  red[threadIdx.x] = s; __syncthreads();
  for (int o = 128; o > 0; o >>= 1) { if (threadIdx.x < o) red[threadIdx.x] += red[threadIdx.x + o]; __syncthreads(); }
  float mean = red[0] * (1.0f / NDIM);
  __syncthreads();
  red[threadIdx.x] = s2; __syncthreads();
  for (int o = 128; o > 0; o >>= 1) { if (threadIdx.x < o) red[threadIdx.x] += red[threadIdx.x + o]; __syncthreads(); }
  float var = red[0] * (1.0f / NDIM) - mean * mean;
  float rstd = rsqrtf(var + 1e-5f);

  float y0 = (v0 - mean) * rstd * g[base + 0] + b[base + 0];
  float y1 = (v1 - mean) * rstd * g[base + 1] + b[base + 1];
  float y2 = (v2 - mean) * rstd * g[base + 2] + b[base + 2];
  float y3 = (v3 - mean) * rstd * g[base + 3] + b[base + 3];
  unsigned int* xo = xn + (size_t)row * (NDIM / 2) + base / 2;
  xo[0] = (unsigned int)f2bf(y0) | ((unsigned int)f2bf(y1) << 16);
  xo[1] = (unsigned int)f2bf(y2) | ((unsigned int)f2bf(y3) << 16);
}

// ------------- Kernel 2: w_qkv transpose + convert (f32 -> bf16) -------------
__global__ __launch_bounds__(256) void wt_kernel(const float* __restrict__ w,
                                                 unsigned short* __restrict__ wT) {
  unsigned int idx = blockIdx.x * 256u + threadIdx.x;   // 0 .. 3072*1024-1
  unsigned int c = idx >> 10;
  unsigned int k = idx & 1023u;
  wT[idx] = f2bf(w[(size_t)k * 3072u + c]);
}

// ---------------- Kernel 3: QKV GEMM via WMMA bf16, async double-buffered ----------------
#define SA2 20                      // 16 dwords (32 bf16 K) + pad, mult of 4
__global__ __launch_bounds__(256) void qkv_gemm(const unsigned int* __restrict__ xn,
                                                const unsigned int* __restrict__ wT,
                                                unsigned short* __restrict__ qbuf,
                                                unsigned short* __restrict__ kbuf,
                                                unsigned short* __restrict__ vtbuf) {
  __shared__ unsigned int As[2 * 128 * SA2];
  __shared__ unsigned int Bs[2 * 128 * SA2];
  int m0 = blockIdx.x * 128;
  int n0 = blockIdx.y * 128;
  int wid = threadIdx.x >> 5, lane = threadIdx.x & 31;
  int waveM = wid >> 1, waveN = wid & 1;          // 4x2 wave grid -> 32x64 per wave
  int lm = lane & 15, lh = lane >> 4;

  AccF acc[2][4];
  for (int mt = 0; mt < 2; ++mt)
    for (int nt = 0; nt < 4; ++nt)
      for (int e = 0; e < 8; ++e) acc[mt][nt].f[e] = 0.f;

  // loader role: threads 0..127 stream one A row each, 128..255 one B row each
  int isB = threadIdx.x >> 7;
  int lrow = threadIdx.x & 127;
  const unsigned int* gsrc = isB ? (wT + (size_t)(n0 + lrow) * 512)
                                 : (xn + (size_t)(m0 + lrow) * 512);
  unsigned int* lbase = (isB ? Bs : As) + lrow * SA2;

#define GEMM_ISSUE(bufi, k0dw) do {                          \
    unsigned int* d = lbase + (bufi) * (128 * SA2);          \
    const unsigned int* sp = gsrc + (k0dw);                  \
    async_b128(d + 0,  sp + 0);  async_b128(d + 4,  sp + 4); \
    async_b128(d + 8,  sp + 8);  async_b128(d + 12, sp + 12);\
  } while (0)

  GEMM_ISSUE(0, 0);
  wait_async0();
  __syncthreads();

  int buf = 0;
  for (int k0 = 0; k0 < 512; k0 += 16) {          // 16 dwords = 32 bf16 K per stage
    if (k0 + 16 < 512) GEMM_ISSUE(buf ^ 1, k0 + 16);

    const unsigned int* Ab = As + buf * (128 * SA2);
    const unsigned int* Bb = Bs + buf * (128 * SA2);
    FragBF af[2], bfr[4];
    for (int mt = 0; mt < 2; ++mt) {
      const uint4* p = (const uint4*)(Ab + (waveM * 32 + mt * 16 + lm) * SA2 + lh * 8);
      af[mt].q[0] = p[0];                         // K = kh*16 .. +7
      af[mt].q[1] = p[1];                         // K = kh*16 + 8 .. +15
    }
    for (int nt = 0; nt < 4; ++nt) {
      const uint4* p = (const uint4*)(Bb + (waveN * 64 + nt * 16 + lm) * SA2 + lh * 8);
      bfr[nt].q[0] = p[0];                        // K = kh*16 .. +15
      bfr[nt].q[1] = p[1];
    }
    for (int mt = 0; mt < 2; ++mt)
      for (int nt = 0; nt < 4; ++nt)
        acc[mt][nt].v = __builtin_amdgcn_wmma_f32_16x16x32_bf16(
            false, af[mt].v, false, bfr[nt].v, (short)0, acc[mt][nt].v, false, false);

    wait_async0();
    __syncthreads();
    buf ^= 1;
  }
#undef GEMM_ISSUE

  // Epilogue: scatter C-layout (lane = col, vgpr e = row e + lh*8) to q/k/vT bf16
  for (int mt = 0; mt < 2; ++mt)
    for (int nt = 0; nt < 4; ++nt) {
      int cg = n0 + waveN * 64 + nt * 16 + lm;    // 0..3071
      int which = cg >> 10;                       // 0=q 1=k 2=v
      int d = cg & 1023;
      int h = d >> 6, dd = d & 63;
      for (int e = 0; e < 8; ++e) {
        int rg = m0 + waveM * 32 + mt * 16 + lh * 8 + e;   // 0..8191
        int bidx = rg >> 11, i = rg & 2047;
        int bh = bidx * HEADS + h;
        unsigned short val = f2bf(acc[mt][nt].f[e]);
        if (which == 0)      qbuf[((size_t)bh * NTOK + i) * DH + dd] = val;
        else if (which == 1) kbuf[((size_t)bh * NTOK + i) * DH + dd] = val;
        else                 vtbuf[((size_t)bh * DH + dd) * NTOK + i] = val;
      }
    }
}

// ---------------- Kernel 4: Flash attention via WMMA bf16, TDM double-buffered ----------------
#define SK 36
__global__ __launch_bounds__(256) void flash_attn(const unsigned int* __restrict__ qbuf,
                                                  const unsigned int* __restrict__ kbuf,
                                                  const unsigned int* __restrict__ vtbuf,
                                                  const int* __restrict__ mask,
                                                  float* __restrict__ out) {
  __shared__ unsigned int Kt[2 * 64 * SK];     // [j][dh]   bf16 pairs, TDM-padded rows
  __shared__ unsigned int Vt[2 * 64 * SK];     // [dd][j]   bf16 pairs, TDM-padded rows
  __shared__ unsigned int Pb[8 * 16 * SK];     // per-wave P tile [16][64] bf16
  __shared__ float maskadd[2][64];

  int bh = blockIdx.y;                     // 0..63
  int bidx = bh >> 4;
  int h = bh & 15;
  int q0 = blockIdx.x * 128;
  int wid = threadIdx.x >> 5, lane = threadIdx.x & 31;
  int lm = lane & 15, lh = lane >> 4;
  int i0 = q0 + wid * 16;

  const unsigned int* qrow = qbuf + (size_t)bh * NTOK * (DH / 2);
  const unsigned int* krow = kbuf + (size_t)bh * NTOK * (DH / 2);
  const unsigned int* vrow = vtbuf + (size_t)bh * DH * (NTOK / 2);
  unsigned long long kgbase = (unsigned long long)(const void*)krow;
  unsigned long long vgbase = (unsigned long long)(const void*)vrow;

  // Q A-fragments (K dim = dh = 64 -> two 16x32 fragments), loaded once
  FragBF qf[2];
  for (int kk = 0; kk < 2; ++kk) {
    const uint4* p = (const uint4*)(qrow + (size_t)(i0 + lm) * 32 + kk * 16 + lh * 8);
    qf[kk].q[0] = p[0];
    qf[kk].q[1] = p[1];
  }

  AccF o[4];
  for (int n = 0; n < 4; ++n)
    for (int e = 0; e < 8; ++e) o[n].f[e] = 0.f;
  float mrow[8], lrow[8];
  for (int e = 0; e < 8; ++e) { mrow[e] = NEG_BIG; lrow[e] = 0.f; }

  unsigned int* Pw = Pb + wid * 16 * SK;

  // TDM issue: wave 0 -> K tile descriptor, wave 1 -> V^T tile descriptor.
  // (TDM ignores EXEC; operands are wave-uniform -> SGPR groups.)
#define ATT_ISSUE(bufi, j0v) do {                                              \
    if (wid == 0) {                                                            \
      unsigned int laddr = (unsigned int)(unsigned long long)(Kt + (bufi) * (64 * SK)); \
      tdm_load_2d(laddr, kgbase + (unsigned long long)(j0v) * (DH * 2),        \
                  DH, NTOK, DH, 64, DH);                                       \
    } else if (wid == 1) {                                                     \
      unsigned int laddr = (unsigned int)(unsigned long long)(Vt + (bufi) * (64 * SK)); \
      tdm_load_2d(laddr, vgbase + (unsigned long long)(j0v) * 2,               \
                  NTOK, DH, 64, DH, NTOK);                                     \
    }                                                                          \
    if (threadIdx.x < 64) {                                                    \
      int mm = mask[(size_t)bidx * NTOK + (j0v) + threadIdx.x];                \
      maskadd[bufi][threadIdx.x] = (mm != 0) ? NEG_BIG : 0.0f;                 \
    }                                                                          \
  } while (0)

  ATT_ISSUE(0, 0);
  if (wid < 2) wait_tensor0();
  __syncthreads();

  int buf = 0;
  for (int j0 = 0; j0 < NTOK; j0 += 64) {
    if (j0 + 64 < NTOK) ATT_ISSUE(buf ^ 1, j0 + 64);

    const unsigned int* Kb = Kt + buf * (64 * SK);
    const unsigned int* Vb = Vt + buf * (64 * SK);

    // S = Q * K^T : 4 column tiles of 16, K-dim 64
    AccF s[4];
    for (int nt = 0; nt < 4; ++nt) {
      for (int e = 0; e < 8; ++e) s[nt].f[e] = 0.f;
      for (int kk = 0; kk < 2; ++kk) {
        FragBF kf;
        const uint4* p = (const uint4*)(Kb + (nt * 16 + lm) * SK + kk * 16 + lh * 8);
        kf.q[0] = p[0]; kf.q[1] = p[1];
        s[nt].v = __builtin_amdgcn_wmma_f32_16x16x32_bf16(
            false, qf[kk].v, false, kf.v, (short)0, s[nt].v, false, false);
      }
      float ma = maskadd[buf][nt * 16 + lm];
      for (int e = 0; e < 8; ++e) s[nt].f[e] = s[nt].f[e] * ATT_SCALE + ma;
    }

    // online softmax row stats (rows live across 16-lane halves)
    float mnew[8], alpha[8];
    for (int e = 0; e < 8; ++e) {
      float mx = s[0].f[e];
      mx = fmaxf(mx, s[1].f[e]); mx = fmaxf(mx, s[2].f[e]); mx = fmaxf(mx, s[3].f[e]);
      mx = half16_max(mx);
      mnew[e] = fmaxf(mrow[e], mx);
      alpha[e] = __expf(mrow[e] - mnew[e]);
      mrow[e] = mnew[e];
    }
    for (int nt = 0; nt < 4; ++nt)
      for (int e = 0; e < 8; ++e) s[nt].f[e] = __expf(s[nt].f[e] - mnew[e]);
    for (int e = 0; e < 8; ++e) {
      float su = s[0].f[e] + s[1].f[e] + s[2].f[e] + s[3].f[e];
      su = half16_sum(su);
      lrow[e] = lrow[e] * alpha[e] + su;
    }
    for (int n = 0; n < 4; ++n)
      for (int e = 0; e < 8; ++e) o[n].f[e] *= alpha[e];

    // C-layout P -> per-wave LDS (bf16) -> A-layout fragments (same wave, no barrier)
    {
      unsigned short* Pws = (unsigned short*)Pw;      // row stride SK*2 shorts
      for (int nt = 0; nt < 4; ++nt)
        for (int e = 0; e < 8; ++e) {
          int row = e + lh * 8;
          int col = nt * 16 + lm;
          Pws[row * (SK * 2) + col] = f2bf(s[nt].f[e]);
        }
    }
    for (int kk2 = 0; kk2 < 2; ++kk2) {
      FragBF pf;
      const uint4* p = (const uint4*)(Pw + lm * SK + kk2 * 16 + lh * 8);
      pf.q[0] = p[0];
      pf.q[1] = p[1];
      for (int n = 0; n < 4; ++n) {
        FragBF vf;
        const uint4* pv = (const uint4*)(Vb + (n * 16 + lm) * SK + kk2 * 16 + lh * 8);
        vf.q[0] = pv[0]; vf.q[1] = pv[1];
        o[n].v = __builtin_amdgcn_wmma_f32_16x16x32_bf16(
            false, pf.v, false, vf.v, (short)0, o[n].v, false, false);
      }
    }

    if (wid < 2) wait_tensor0();
    __syncthreads();
    buf ^= 1;
  }
#undef ATT_ISSUE

  // normalize + write out[b][i][h*64+dd] in f32
  for (int e = 0; e < 8; ++e) {
    float inv = 1.0f / lrow[e];
    int i = i0 + e + lh * 8;
    size_t base = ((size_t)bidx * NTOK + i) * NDIM + h * DH;
    for (int n = 0; n < 4; ++n)
      out[base + n * 16 + lm] = o[n].f[e] * inv;
  }
}

extern "C" void kernel_launch(void* const* d_in, const int* in_sizes, int n_in,
                              void* d_out, int out_size, void* d_ws, size_t ws_size,
                              hipStream_t stream) {
  const float* x    = (const float*)d_in[0];
  const int*   mask = (const int*)d_in[1];
  const float* wqkv = (const float*)d_in[2];
  const float* ln_g = (const float*)d_in[3];
  const float* ln_b = (const float*)d_in[4];
  float* out = (float*)d_out;

  char* ws = (char*)d_ws;
  size_t off = 0;
  unsigned int*   xn  = (unsigned int*)(ws + off); off += (size_t)8192 * 512 * 4;    // 16.8 MB
  unsigned int*   wT  = (unsigned int*)(ws + off); off += (size_t)3072 * 512 * 4;    //  6.3 MB
  unsigned short* qb  = (unsigned short*)(ws + off); off += (size_t)64 * NTOK * DH * 2;
  unsigned short* kb  = (unsigned short*)(ws + off); off += (size_t)64 * NTOK * DH * 2;
  unsigned short* vtb = (unsigned short*)(ws + off); off += (size_t)64 * NTOK * DH * 2;

  ln_kernel<<<8192, 256, 0, stream>>>(x, ln_g, ln_b, xn);
  wt_kernel<<<(3072 * 1024) / 256, 256, 0, stream>>>(wqkv, (unsigned short*)wT);
  qkv_gemm<<<dim3(64, 24), 256, 0, stream>>>(xn, wT, qb, kb, vtb);
  flash_attn<<<dim3(16, 64), 256, 0, stream>>>((const unsigned int*)qb,
                                               (const unsigned int*)kb,
                                               (const unsigned int*)vtb,
                                               mask, out);
}
